// OptimizedGNN_43430709297250
// MI455X (gfx1250) — compile-verified, matrix operand order checked by reading, not run on hardware
//
#include <hip/hip_runtime.h>
#include <hip/hip_bf16.h>

typedef __attribute__((ext_vector_type(16))) _Float16 v16h;
typedef __attribute__((ext_vector_type(8)))  float    v8f;

#define HEADS   4
#define OUT_CH  64
#define HC      256      // HEADS*OUT_CH
#define IN_CH   256
#define POS_DIM 16
#define XSTRIDE 272      // IN_CH + POS_DIM

// ---------------------------------------------------------------------------
// Kernel 1: init out=bias (self-loop aggregation lands on top), amax=-FLT_MAX,
// denom=0, G=0. Deterministic re-init every launch.
// ---------------------------------------------------------------------------
__global__ void k_init(float* __restrict__ out, const float* __restrict__ bias,
                       float* __restrict__ amax, float* __restrict__ denom,
                       float* __restrict__ G, int N) {
    int i = blockIdx.x * blockDim.x + threadIdx.x;
    if (i < N * HC) out[i] = bias[i & (HC - 1)];
    if (i < N * HEADS) { amax[i] = -3.402823466e38f; denom[i] = 0.0f; }
    if (i < HEADS * HEADS) G[i] = 0.0f;
}

// ---------------------------------------------------------------------------
// Kernel 2: content[N,256] = x[:, :256] @ lin_w.T  via v_wmma_f32_16x16x32_f16
// Block = 4 waves; wave w computes a 16x64 strip (4 accumulators, n0 = w*64),
// reusing its A fragment across 4 WMMAs per k-step (A is the HBM-streamed
// operand; lin_w is L2-resident). Grid = N/16 m-tiles.
// A lane layout (16-bit, 16x32): lane<16 -> K 0..7 (dw0..3), 16..23 (dw4..7);
// lane>=16 -> K 8..15, 24..31.  B mirrors it with n = lane&15.
// ---------------------------------------------------------------------------
__global__ __launch_bounds__(128) void k_gemm_content(
        const float* __restrict__ x, const float* __restrict__ w,
        float* __restrict__ content) {
    const int lane   = threadIdx.x & 31;
    const int wave   = threadIdx.x >> 5;
    const int m0     = blockIdx.x * 16;
    const int n0     = wave * 64;          // 4 n-tiles per wave
    const int half16 = lane >> 4;          // 0 or 1
    const int idx16  = lane & 15;
    const int khalf  = half16 * 8;

    const float* arp = x + (size_t)(m0 + idx16) * XSTRIDE;  // A row (M)
    const float* brp0 = w + (size_t)(n0 + idx16) * IN_CH;   // lin_w rows (out ch)

    v8f acc0 = {}, acc1 = {}, acc2 = {}, acc3 = {};
#pragma unroll
    for (int k0 = 0; k0 < IN_CH; k0 += 32) {
        // ---- A fragment: loaded once, reused by 4 WMMAs ----
        v16h a;
#pragma unroll
        for (int j = 0; j < 8; ++j) {
            const int kk = k0 + khalf + ((j < 4) ? (2 * j) : (16 + 2 * (j - 4)));
            const float2 av = *(const float2*)(arp + kk);
            a[2 * j]     = (_Float16)av.x;
            a[2 * j + 1] = (_Float16)av.y;
        }
        // ---- 4 B fragments / 4 WMMAs ----
#pragma unroll
        for (int t = 0; t < 4; ++t) {
            const float* brp = brp0 + (size_t)t * 16 * IN_CH;
            v16h b;
#pragma unroll
            for (int j = 0; j < 8; ++j) {
                const int kk = k0 + khalf + ((j < 4) ? (2 * j) : (16 + 2 * (j - 4)));
                const float2 bv = *(const float2*)(brp + kk);
                b[2 * j]     = (_Float16)bv.x;
                b[2 * j + 1] = (_Float16)bv.y;
            }
            v8f* accp = (t == 0) ? &acc0 : (t == 1) ? &acc1 : (t == 2) ? &acc2 : &acc3;
            *accp = __builtin_amdgcn_wmma_f32_16x16x32_f16(
                        /*neg_a=*/false, a, /*neg_b=*/false, b,
                        /*c_mod=*/(short)0, *accp,
                        /*reuse_a=*/false, /*reuse_b=*/false);
        }
    }
    // C/D layout: lane<16 -> M=j, lane>=16 -> M=j+8 ; N = idx16
    float* cbase = content + (size_t)(m0 + half16 * 8) * HC + n0 + idx16;
#pragma unroll
    for (int t = 0; t < 4; ++t) {
        const v8f acc = (t == 0) ? acc0 : (t == 1) ? acc1 : (t == 2) ? acc2 : acc3;
        float* cp = cbase + t * 16;
#pragma unroll
        for (int j = 0; j < 8; ++j) cp[(size_t)j * HC] = acc[j];
    }
}

// ---------------------------------------------------------------------------
// Kernel 3: per-(node,head) attention terms.
// srcterm = a_src + p_src ; dstterm = a_dst + p_dst
// ---------------------------------------------------------------------------
__global__ void k_node_terms(const float* __restrict__ x,
                             const float* __restrict__ content,
                             const float* __restrict__ att,
                             const float* __restrict__ pos_att,
                             float* __restrict__ srcterm,
                             float* __restrict__ dstterm, int N) {
    int t = blockIdx.x * blockDim.x + threadIdx.x;
    if (t >= N * HEADS) return;
    const int n = t >> 2, h = t & 3;
    const float* crow = content + (size_t)n * HC + h * OUT_CH;
    const float* arow = att + h * (2 * OUT_CH);
    float as = 0.f, ad = 0.f;
#pragma unroll 4
    for (int cIdx = 0; cIdx < OUT_CH; ++cIdx) {
        const float v = crow[cIdx];
        as += v * arow[cIdx];
        ad += v * arow[OUT_CH + cIdx];
    }
    const float* prow  = x + (size_t)n * XSTRIDE + IN_CH;
    const float* parow = pos_att + h * (2 * POS_DIM);
    float ps = 0.f, pd = 0.f;
#pragma unroll
    for (int d = 0; d < POS_DIM; ++d) {
        const float p = prow[d];
        ps += p * parow[d];
        pd += p * parow[POS_DIM + d];
    }
    srcterm[t] = as + ps;
    dstterm[t] = ad + pd;
}

// float atomic max via int/uint punning (valid with -FLT_MAX init)
__device__ __forceinline__ void atomicMaxF(float* addr, float val) {
    if (val >= 0.0f) atomicMax((int*)addr, __float_as_int(val));
    else             atomicMin((unsigned int*)addr, __float_as_uint(val));
}

__device__ __forceinline__ float edge_logit(const float* __restrict__ srcterm,
                                            const float* __restrict__ dstterm,
                                            int r, int cidx, int h) {
    float a = srcterm[r * HEADS + h] + dstterm[cidx * HEADS + h];
    return (a > 0.0f) ? a : a * 0.2f;   // leaky_relu(0.2)
}

// ---------------------------------------------------------------------------
// Kernel 4: segment max over rows (thread per (edge,head), incl. self-loops)
// ---------------------------------------------------------------------------
__global__ void k_edge_max(const int* __restrict__ row, const int* __restrict__ col,
                           const float* __restrict__ srcterm,
                           const float* __restrict__ dstterm,
                           float* __restrict__ amax, int E, int N) {
    int t = blockIdx.x * blockDim.x + threadIdx.x;
    const int Etot = E + N;
    if (t >= Etot * HEADS) return;
    const int e = t >> 2, h = t & 3;
    const int r    = (e < E) ? row[e] : (e - E);
    const int cidx = (e < E) ? col[e] : (e - E);
    atomicMaxF(&amax[r * HEADS + h], edge_logit(srcterm, dstterm, r, cidx, h));
}

// ---------------------------------------------------------------------------
// Kernel 5: segment sum of exp(logit - amax[row])
// ---------------------------------------------------------------------------
__global__ void k_edge_denom(const int* __restrict__ row, const int* __restrict__ col,
                             const float* __restrict__ srcterm,
                             const float* __restrict__ dstterm,
                             const float* __restrict__ amax,
                             float* __restrict__ denom, int E, int N) {
    int t = blockIdx.x * blockDim.x + threadIdx.x;
    const int Etot = E + N;
    if (t >= Etot * HEADS) return;
    const int e = t >> 2, h = t & 3;
    const int r    = (e < E) ? row[e] : (e - E);
    const int cidx = (e < E) ? col[e] : (e - E);
    const float a = edge_logit(srcterm, dstterm, r, cidx, h);
    atomicAdd(&denom[r * HEADS + h], __expf(a - amax[r * HEADS + h]));
}

// ---------------------------------------------------------------------------
// Kernel 6: one wave per edge: out[row] += content[col] * alpha  (coalesced
// gather + atomic scatter over 256 channels), plus Gram-matrix accumulation
// for the diversity loss (block-shared reduce -> 16 global atomics).
// ---------------------------------------------------------------------------
__global__ __launch_bounds__(256) void k_edge_scatter(
        const int* __restrict__ row, const int* __restrict__ col,
        const float* __restrict__ srcterm, const float* __restrict__ dstterm,
        const float* __restrict__ amax, const float* __restrict__ denom,
        const float* __restrict__ content, float* __restrict__ out,
        float* __restrict__ G, int E, int N) {
    __shared__ float sG[HEADS * HEADS];
    if (threadIdx.x < HEADS * HEADS) sG[threadIdx.x] = 0.0f;
    __syncthreads();

    const int wave = threadIdx.x >> 5;
    const int lane = threadIdx.x & 31;
    const long long e = (long long)blockIdx.x * 8 + wave;
    const int Etot = E + N;

    if (e < Etot) {
        const int r    = (e < E) ? row[e] : (int)(e - E);
        const int cidx = (e < E) ? col[e] : (int)(e - E);
        float al[HEADS];
#pragma unroll
        for (int h = 0; h < HEADS; ++h) {
            const float a = edge_logit(srcterm, dstterm, r, cidx, h);
            al[h] = __expf(a - amax[r * HEADS + h]) /
                    (denom[r * HEADS + h] + 1e-16f);
        }
        if (lane == 0) {
#pragma unroll
            for (int h1 = 0; h1 < HEADS; ++h1)
#pragma unroll
                for (int h2 = 0; h2 < HEADS; ++h2)
                    atomicAdd(&sG[h1 * HEADS + h2], al[h1] * al[h2]);
        }
        const float* crow = content + (size_t)cidx * HC;
        float*       orow = out     + (size_t)r    * HC;
#pragma unroll
        for (int i = 0; i < 8; ++i) {
            const int ch = lane + 32 * i;
            atomicAdd(&orow[ch], crow[ch] * al[ch >> 6]);
        }
    }
    __syncthreads();
    if (threadIdx.x < HEADS * HEADS) atomicAdd(&G[threadIdx.x], sG[threadIdx.x]);
}

// ---------------------------------------------------------------------------
// Kernel 7: diversity loss from the 4x4 Gram matrix.
// norms = max(sqrt(diag(G)), 1e-8); corr off-diag = G/(n1*n2); mean*0.1
// ---------------------------------------------------------------------------
__global__ void k_diversity(const float* __restrict__ G, float* __restrict__ loss) {
    if (threadIdx.x == 0 && blockIdx.x == 0) {
        float norms[HEADS];
#pragma unroll
        for (int h = 0; h < HEADS; ++h)
            norms[h] = fmaxf(sqrtf(G[h * HEADS + h]), 1e-8f);
        float s = 0.0f;
#pragma unroll
        for (int h1 = 0; h1 < HEADS; ++h1)
#pragma unroll
            for (int h2 = 0; h2 < HEADS; ++h2)
                if (h1 != h2) s += G[h1 * HEADS + h2] / (norms[h1] * norms[h2]);
        loss[0] = (s / (float)(HEADS * HEADS)) * 0.1f;
    }
}

// ---------------------------------------------------------------------------
extern "C" void kernel_launch(void* const* d_in, const int* in_sizes, int n_in,
                              void* d_out, int out_size, void* d_ws, size_t ws_size,
                              hipStream_t stream) {
    const float* x       = (const float*)d_in[0];   // [N, 272]
    const int*   edges   = (const int*)  d_in[1];   // [2, E]
    const float* lin_w   = (const float*)d_in[2];   // [256, 256]
    const float* att     = (const float*)d_in[3];   // [1, 4, 128]
    const float* pos_att = (const float*)d_in[4];   // [1, 4, 32]
    const float* bias    = (const float*)d_in[5];   // [256]

    const int N = in_sizes[0] / XSTRIDE;            // 50000
    const int E = in_sizes[1] / 2;                  // 500000
    const int* row = edges;
    const int* col = edges + E;
    const int Etot = E + N;

    // workspace layout (floats)
    float* content = (float*)d_ws;                       // N*256
    float* srcterm = content + (size_t)N * HC;           // N*4
    float* dstterm = srcterm + (size_t)N * HEADS;        // N*4
    float* amax    = dstterm + (size_t)N * HEADS;        // N*4
    float* denom   = amax    + (size_t)N * HEADS;        // N*4
    float* G       = denom   + (size_t)N * HEADS;        // 16

    float* out  = (float*)d_out;                         // N*256
    float* loss = out + (size_t)N * HC;                  // +1

    // 1) init out=bias, amax=-inf, denom=0, G=0
    {
        int total = N * HC;
        k_init<<<(total + 255) / 256, 256, 0, stream>>>(out, bias, amax, denom, G, N);
    }
    // 2) WMMA projection: content = x[:, :256] @ lin_w.T  (16x64 strip/wave)
    {
        k_gemm_content<<<N / 16, 128, 0, stream>>>(x, lin_w, content);
    }
    // 3) per-node attention terms
    {
        int total = N * HEADS;
        k_node_terms<<<(total + 255) / 256, 256, 0, stream>>>(
            x, content, att, pos_att, srcterm, dstterm, N);
    }
    // 4) segment max
    {
        int total = Etot * HEADS;
        k_edge_max<<<(total + 255) / 256, 256, 0, stream>>>(
            row, col, srcterm, dstterm, amax, E, N);
    }
    // 5) segment sum of exp
    {
        int total = Etot * HEADS;
        k_edge_denom<<<(total + 255) / 256, 256, 0, stream>>>(
            row, col, srcterm, dstterm, amax, denom, E, N);
    }
    // 6) message scatter + Gram accumulation (wave per edge)
    {
        int blocks = (Etot + 7) / 8;
        k_edge_scatter<<<blocks, 256, 0, stream>>>(
            row, col, srcterm, dstterm, amax, denom, content, out, G, E, N);
    }
    // 7) diversity loss
    k_diversity<<<1, 32, 0, stream>>>(G, loss);
}
